// WindowAttention1D_86045374808600
// MI455X (gfx1250) — matrix operand order repaired and checked by reading, not compiled, not run on hardware
//
#include <hip/hip_runtime.h>
#include <hip/hip_bf16.h>
#include <stdint.h>

// ---------------------------------------------------------------------------
// Types for WMMA (gfx1250 / CDNA5, wave32)
// ---------------------------------------------------------------------------
typedef __bf16  v16bf  __attribute__((ext_vector_type(16)));
typedef __bf16  bf16x8 __attribute__((ext_vector_type(8)));
typedef __bf16  bf16x4 __attribute__((ext_vector_type(4)));
typedef float   v8f    __attribute__((ext_vector_type(8)));

union FragU {
    v16bf  v;
    bf16x8 h[2];
};

__device__ __forceinline__ v8f wmma_bf16(const FragU& a, const FragU& b, v8f c) {
    return __builtin_amdgcn_wmma_f32_16x16x32_bf16(
        /*neg_a=*/false, a.v, /*neg_b=*/false, b.v,
        /*c_mod=*/(short)0, c, /*reuse_a=*/false, /*reuse_b=*/false);
}

#define ZERO8 {0.f,0.f,0.f,0.f,0.f,0.f,0.f,0.f}

// Problem constants
#define KDIM   512
#define NQKV   1536
#define HEADS  8
#define DH     64
#define WIN    64
#define NWIN   64
#define BATCH  8
#define MROWS  32768   // B*NW*W

// ---------------------------------------------------------------------------
// Pre-pass 1: X f32 -> bf16, same row-major layout. 8 elems/thread.
// ---------------------------------------------------------------------------
__global__ __launch_bounds__(256)
void wa_cvt_x(const float* __restrict__ X, __bf16* __restrict__ Xb)
{
    size_t i = ((size_t)blockIdx.x * 256 + threadIdx.x) * 8;
    float4 f0 = *(const float4*)(X + i);
    float4 f1 = *(const float4*)(X + i + 4);
    bf16x8 h;
    h[0] = (__bf16)f0.x; h[1] = (__bf16)f0.y; h[2] = (__bf16)f0.z; h[3] = (__bf16)f0.w;
    h[4] = (__bf16)f1.x; h[5] = (__bf16)f1.y; h[6] = (__bf16)f1.z; h[7] = (__bf16)f1.w;
    *(bf16x8*)(Xb + i) = h;
}

// ---------------------------------------------------------------------------
// Pre-pass 2: W[512][N] f32 -> Wt[N][512] bf16 (transpose + convert), so the
// GEMM's B operand is K-contiguous and hot-loop staging is pure b128 copies.
// 32x32 tiles through LDS.
// ---------------------------------------------------------------------------
__global__ __launch_bounds__(256)
void wa_transpose_w(const float* __restrict__ W, __bf16* __restrict__ Wt, int N)
{
    __shared__ __align__(16) __bf16 t[32 * 36];
    const int k0  = blockIdx.x * 32;
    const int n0  = blockIdx.y * 32;
    const int tid = threadIdx.x;
    {
        int r  = tid >> 3;            // k within tile
        int c4 = (tid & 7) * 4;       // n within tile
        float4 f = *(const float4*)(W + (size_t)(k0 + r) * N + n0 + c4);
        bf16x4 h;
        h[0] = (__bf16)f.x; h[1] = (__bf16)f.y; h[2] = (__bf16)f.z; h[3] = (__bf16)f.w;
        *(bf16x4*)&t[r * 36 + c4] = h;
    }
    __syncthreads();
    {
        int n  = tid >> 3;
        int k4 = (tid & 7) * 4;
        bf16x4 h;
#pragma unroll
        for (int j = 0; j < 4; j++) h[j] = t[(k4 + j) * 36 + n];
        *(bf16x4*)(Wt + (size_t)(n0 + n) * KDIM + k0 + k4) = h;
    }
}

// ---------------------------------------------------------------------------
// GEMM: C[M,LDC-slice] = A(bf16)[M,512] x Bt(bf16)[N,512]^T
// Block 256 thr (8 waves), tile 128x128, K-step 32, double-buffered LDS.
// Both operands K-contiguous in LDS, stride 40 (16B-aligned, conflict-free:
// 20*r mod 64 injective over 16 rows). Fragments = 2x ds_load_b128 each.
// PROJ=true: f32 output + bias; PROJ=false: bf16 output.
// ---------------------------------------------------------------------------
template<int LDC, bool PROJ>
__global__ __launch_bounds__(256)
void wa_gemm(const __bf16* __restrict__ A, const __bf16* __restrict__ Bt,
             const float* __restrict__ bias, void* __restrict__ Cv)
{
    __shared__ __align__(16) __bf16 sA[2][128 * 40];
    __shared__ __align__(16) __bf16 sB[2][128 * 40];

    const int tid  = threadIdx.x;
    const int row0 = blockIdx.x * 128;
    const int col0 = blockIdx.y * 128;
    const int lane = tid & 31;
    const int wid  = tid >> 5;
    const int wm   = wid >> 2;      // 0..1 -> 64-row strip
    const int wn   = wid & 3;       // 0..3 -> 32-col strip
    const int lr   = lane & 15;
    const int hi   = lane >> 4;

    // staging pattern (identical for A and Bt): 128 rows x 32 k bf16
    const int sr0 = tid >> 2;             // it=0 row
    const int sc  = (tid & 3) * 8;        // chunk offset in k
    const int sr1 = sr0 + 64;             // it=1 row

    bf16x8 ra0, ra1, rb0, rb1;
    auto stage_load = [&](int kk) {
        ra0 = *(const bf16x8*)(A  + (size_t)(row0 + sr0) * KDIM + kk + sc);
        ra1 = *(const bf16x8*)(A  + (size_t)(row0 + sr1) * KDIM + kk + sc);
        rb0 = *(const bf16x8*)(Bt + (size_t)(col0 + sr0) * KDIM + kk + sc);
        rb1 = *(const bf16x8*)(Bt + (size_t)(col0 + sr1) * KDIM + kk + sc);
    };
    auto stage_store = [&](int buf) {
        *(bf16x8*)&sA[buf][sr0 * 40 + sc] = ra0;
        *(bf16x8*)&sA[buf][sr1 * 40 + sc] = ra1;
        *(bf16x8*)&sB[buf][sr0 * 40 + sc] = rb0;
        *(bf16x8*)&sB[buf][sr1 * 40 + sc] = rb1;
    };

    v8f acc[4][2];
#pragma unroll
    for (int i = 0; i < 4; i++)
#pragma unroll
        for (int j = 0; j < 2; j++) { v8f z = ZERO8; acc[i][j] = z; }

    stage_load(0);
    stage_store(0);
    __syncthreads();

    for (int s = 0; s < KDIM / 32; ++s) {
        const int cur = s & 1;
        if (s + 1 < KDIM / 32) stage_load((s + 1) * 32);

        FragU a[4], b[2];
#pragma unroll
        for (int mt = 0; mt < 4; mt++) {
            int r = wm * 64 + mt * 16 + lr;
            a[mt].h[0] = *(const bf16x8*)&sA[cur][r * 40 + hi * 8];       // k hi*8..+7
            a[mt].h[1] = *(const bf16x8*)&sA[cur][r * 40 + 16 + hi * 8];  // k 16+hi*8..
        }
#pragma unroll
        for (int nt = 0; nt < 2; nt++) {
            int n = wn * 32 + nt * 16 + lr;
            b[nt].h[0] = *(const bf16x8*)&sB[cur][n * 40 + hi * 8];       // k hi*8..+7
            b[nt].h[1] = *(const bf16x8*)&sB[cur][n * 40 + 16 + hi * 8];  // k 16+hi*8..
        }
#pragma unroll
        for (int mt = 0; mt < 4; mt++)
#pragma unroll
            for (int nt = 0; nt < 2; nt++)
                acc[mt][nt] = wmma_bf16(a[mt], b[nt], acc[mt][nt]);

        if (s + 1 < KDIM / 32) stage_store(cur ^ 1);
        __syncthreads();
    }

#pragma unroll
    for (int mt = 0; mt < 4; mt++)
#pragma unroll
        for (int nt = 0; nt < 2; nt++)
#pragma unroll
            for (int i = 0; i < 8; i++) {
                int r = row0 + wm * 64 + mt * 16 + hi * 8 + i;
                int c = col0 + wn * 32 + nt * 16 + lr;
                if constexpr (PROJ) {
                    ((float*)Cv)[(size_t)r * LDC + c] = acc[mt][nt][i] + bias[c];
                } else {
                    ((__bf16*)Cv)[(size_t)r * LDC + c] = (__bf16)acc[mt][nt][i];
                }
            }
}

// ---------------------------------------------------------------------------
// Per-window attention. Block=128 thr (4 waves), one (b,h,window).
// S = Q K^T ; P = softmax(S*0.125) ; O = P V.  All matmuls via WMMA bf16.
// ---------------------------------------------------------------------------
__global__ __launch_bounds__(128)
void wa_attn(const __bf16* __restrict__ QKV, __bf16* __restrict__ O)
{
    __shared__ __align__(16) __bf16 sQ [WIN * 72];
    __shared__ __align__(16) __bf16 sK [WIN * 72];
    __shared__ __align__(16) __bf16 sVt[WIN * 72];
    __shared__ __align__(16) __bf16 sP [WIN * 72];

    const int tid  = threadIdx.x;
    const int lane = tid & 31;
    const int wid  = tid >> 5;         // 0..3: 16-query strip
    const int lr   = lane & 15;
    const int hi   = lane >> 4;

    const int blk = blockIdx.x;        // 0..4095
    const int b   = blk >> 9;          // / (HEADS*NWIN)
    const int rem = blk & 511;
    const int h   = rem >> 6;
    const int n   = rem & 63;

    const size_t m0 = ((size_t)(b * NWIN + n)) * WIN;          // first qkv row
    const __bf16* base = QKV + m0 * NQKV + h * DH;

    // ---- stage Q, K row-major; V transposed (B operand of P*V needs V^T)
#pragma unroll
    for (int it = 0; it < 4; ++it) {
        int idx = it * 128 + tid;       // 0..511
        int r   = idx >> 3;             // 0..63
        int c8  = (idx & 7) * 8;        // 0..56
        bf16x8 q = *(const bf16x8*)(base + (size_t)r * NQKV + c8);
        bf16x8 k = *(const bf16x8*)(base + (size_t)r * NQKV + 512 + c8);
        bf16x8 v = *(const bf16x8*)(base + (size_t)r * NQKV + 1024 + c8);
        *(bf16x8*)&sQ[r * 72 + c8] = q;
        *(bf16x8*)&sK[r * 72 + c8] = k;
#pragma unroll
        for (int j = 0; j < 8; j++) sVt[(c8 + j) * 72 + r] = v[j];
    }
    __syncthreads();

    // ---- S = Q K^T  (wave computes 16x64 strip; 4 key tiles, d-steps of 32)
    v8f s[4];
#pragma unroll
    for (int nt = 0; nt < 4; nt++) { v8f z = ZERO8; s[nt] = z; }
    const int q = wid * 16 + lr;
#pragma unroll
    for (int kb = 0; kb < DH; kb += 32) {
        FragU a;
        a.h[0] = *(const bf16x8*)&sQ[q * 72 + kb + hi * 8];
        a.h[1] = *(const bf16x8*)&sQ[q * 72 + kb + 16 + hi * 8];
#pragma unroll
        for (int nt = 0; nt < 4; nt++) {
            FragU bb;
            int key = nt * 16 + lr;     // B lane = key column; table-order k = K's d
            bb.h[0] = *(const bf16x8*)&sK[key * 72 + kb + hi * 8];
            bb.h[1] = *(const bf16x8*)&sK[key * 72 + kb + 16 + hi * 8];
            s[nt] = wmma_bf16(a, bb, s[nt]);
        }
    }

    // ---- softmax over keys. Lane holds rows wid*16+hi*8+i, cols nt*16+lr.
    // Fully unrolled so all v8f subscripts are constants (no v_movrels).
    const float scale = 0.125f;         // dh^-0.5
#pragma unroll
    for (int i = 0; i < 8; i++) {
        float mx = -3.0e38f;
#pragma unroll
        for (int nt = 0; nt < 4; nt++) mx = fmaxf(mx, s[nt][i]);
#pragma unroll
        for (int off = 1; off < 16; off <<= 1)
            mx = fmaxf(mx, __shfl_xor(mx, off, 32));
        float e[4], sum = 0.f;
#pragma unroll
        for (int nt = 0; nt < 4; nt++) {
            e[nt] = __expf((s[nt][i] - mx) * scale);
            sum += e[nt];
        }
#pragma unroll
        for (int off = 1; off < 16; off <<= 1)
            sum += __shfl_xor(sum, off, 32);
        float inv = 1.0f / sum;
        int qr = wid * 16 + hi * 8 + i;
#pragma unroll
        for (int nt = 0; nt < 4; nt++)
            sP[qr * 72 + nt * 16 + lr] = (__bf16)(e[nt] * inv);
    }
    __syncthreads();

    // ---- O = P V  (A = P rows from LDS, B = V^T columns)
    v8f o[4];
#pragma unroll
    for (int dt = 0; dt < 4; dt++) { v8f z = ZERO8; o[dt] = z; }
#pragma unroll
    for (int kb = 0; kb < WIN; kb += 32) {
        FragU a;
        a.h[0] = *(const bf16x8*)&sP[q * 72 + kb + hi * 8];
        a.h[1] = *(const bf16x8*)&sP[q * 72 + kb + 16 + hi * 8];
#pragma unroll
        for (int dt = 0; dt < 4; dt++) {
            FragU bb;
            int d = dt * 16 + lr;
            bb.h[0] = *(const bf16x8*)&sVt[d * 72 + kb + hi * 8];
            bb.h[1] = *(const bf16x8*)&sVt[d * 72 + kb + 16 + hi * 8];
            o[dt] = wmma_bf16(a, bb, o[dt]);
        }
    }

    // ---- write flat [B,H,nW,W,dh] (= the reference's raw reshape order)
    const size_t obase = ((((size_t)b * HEADS + h) * NWIN + n) * WIN) * DH;
#pragma unroll
    for (int dt = 0; dt < 4; dt++)
#pragma unroll
        for (int i = 0; i < 8; i++) {
            int qr = wid * 16 + hi * 8 + i;
            int d  = dt * 16 + lr;
            O[obase + (size_t)qr * DH + d] = (__bf16)o[dt][i];
        }
}

// ---------------------------------------------------------------------------
// Host launcher
// ---------------------------------------------------------------------------
extern "C" void kernel_launch(void* const* d_in, const int* in_sizes, int n_in,
                              void* d_out, int out_size, void* d_ws, size_t ws_size,
                              hipStream_t stream)
{
    const float* x      = (const float*)d_in[0];   // [8,64,64,512]
    const float* w_qkv  = (const float*)d_in[1];   // [512,1536]
    const float* w_proj = (const float*)d_in[2];   // [512,512]
    const float* b_proj = (const float*)d_in[3];   // [512]
    float*       out    = (float*)d_out;           // [32768,512]

    // workspace: qkv 96MB | attn 32MB | x_bf16 32MB | wqkv_t 1.5MB | wproj_t 0.5MB
    __bf16* qkv_ws  = (__bf16*)d_ws;
    __bf16* attn_ws = qkv_ws  + (size_t)MROWS * NQKV;
    __bf16* xb_ws   = attn_ws + (size_t)MROWS * KDIM;
    __bf16* wqkvt   = xb_ws   + (size_t)MROWS * KDIM;
    __bf16* wprojt  = wqkvt   + (size_t)NQKV  * KDIM;

    // 0) one-time conversions (bf16 + weight transposes)
    wa_cvt_x<<<dim3((MROWS * KDIM) / (256 * 8)), 256, 0, stream>>>(x, xb_ws);
    wa_transpose_w<<<dim3(KDIM / 32, NQKV / 32), 256, 0, stream>>>(w_qkv, wqkvt, NQKV);
    wa_transpose_w<<<dim3(KDIM / 32, KDIM / 32), 256, 0, stream>>>(w_proj, wprojt, KDIM);

    // 1) QKV projection: [32768,512] x [512,1536] -> bf16
    wa_gemm<NQKV, false><<<dim3(MROWS / 128, NQKV / 128), 256, 0, stream>>>(
        xb_ws, wqkvt, nullptr, qkv_ws);

    // 2) windowed attention: 8*8*64 = 4096 (batch, head, window) blocks
    wa_attn<<<dim3(BATCH * HEADS * NWIN), 128, 0, stream>>>(qkv_ws, attn_ws);

    // 3) output projection + bias: [32768,512] x [512,512] -> f32
    wa_gemm<KDIM, true><<<dim3(MROWS / 128, KDIM / 128), 256, 0, stream>>>(
        attn_ws, wprojt, b_proj, out);
}
